// Features2Features_51256139710520
// MI455X (gfx1250) — compile-verified, matrix operand order checked by reading, not run on hardware
//
#include <hip/hip_runtime.h>
#include <hip/hip_bf16.h>

#define NND   100000
#define NEDG  1600000
#define F     128
#define EPSV  1e-5f

typedef __bf16 bf16_t;
typedef bf16_t v16bf __attribute__((ext_vector_type(16)));
typedef bf16_t v8bf  __attribute__((ext_vector_type(8)));
typedef float  v8f   __attribute__((ext_vector_type(8)));

// ---------------------------------------------------------------------------
// float atomic add (lowers to global_atomic_add_f32 on CDNA)
// ---------------------------------------------------------------------------
__device__ __forceinline__ void atomAddF(float* p, float v) {
    unsafeAtomicAdd(p, v);
}

// ---------------------------------------------------------------------------
// Pack a 128x128 row-major f32 weight into bf16 WMMA B-fragments:
// Wp[ct(8)][kb(4)][lane(32)][i(16)]  (one v16bf per lane per k-block)
//   n      = ct*16 + (lane&15)
//   khalf  = lane>>4
//   K(i)   = kb*32 + (i<8 ? khalf*8+i : 16 + khalf*8 + (i-8))
// ---------------------------------------------------------------------------
__global__ __launch_bounds__(256) void pack_w_kernel(const float* __restrict__ W,
                                                     bf16_t* __restrict__ Wp) {
    int o = blockIdx.x * blockDim.x + threadIdx.x;      // 0 .. 16383
    if (o >= F * F) return;
    int i    = o & 15;
    int lane = (o >> 4) & 31;
    int kb   = (o >> 9) & 3;
    int ct   = o >> 11;
    int n     = ct * 16 + (lane & 15);
    int khalf = lane >> 4;
    int k     = kb * 32 + ((i < 8) ? (khalf * 8 + i) : (16 + khalf * 8 + (i - 8)));
    Wp[o] = (bf16_t)W[k * F + n];
}

// ---------------------------------------------------------------------------
// f32 -> bf16 convert (initial features)
// ---------------------------------------------------------------------------
__global__ __launch_bounds__(256) void cvt_kernel(const float* __restrict__ src,
                                                  bf16_t* __restrict__ dst, int n) {
    int idx = blockIdx.x * blockDim.x + threadIdx.x;
    if (idx < n) dst[idx] = (bf16_t)src[idx];
}

// ---------------------------------------------------------------------------
// Dual GEMM: y = xb @ W0 + b0 ; m = xb @ W1 + b1     (bf16 WMMA, f32 accum)
// block = 256 threads = 8 waves; wave ct owns columns [16ct,16ct+16);
// block row-tile = 16 rows (100000 % 16 == 0 -> no tail handling needed).
// ---------------------------------------------------------------------------
__global__ __launch_bounds__(256) void gemm_dual_kernel(
    const bf16_t* __restrict__ xb,
    const bf16_t* __restrict__ wp0, const float* __restrict__ b0,
    const bf16_t* __restrict__ wp1, const float* __restrict__ b1,
    float* __restrict__ y, float* __restrict__ m)
{
    const int lane    = threadIdx.x & 31;
    const int ct      = threadIdx.x >> 5;          // 0..7 column tile
    const int rowbase = blockIdx.x * 16;
    const int khalf   = lane >> 4;
    const int rowA    = rowbase + (lane & 15);

    v8f acc0 = {};
    v8f acc1 = {};
    const bf16_t* arow = xb + (size_t)rowA * F;

#pragma unroll
    for (int kb = 0; kb < 4; ++kb) {
        const bf16_t* ap = arow + kb * 32 + khalf * 8;
        v8bf lo = *(const v8bf*)(ap);
        v8bf hi = *(const v8bf*)(ap + 16);
        v16bf a = __builtin_shufflevector(lo, hi,
                    0, 1, 2, 3, 4, 5, 6, 7, 8, 9, 10, 11, 12, 13, 14, 15);
        v16bf bw0 = *(const v16bf*)(wp0 + (((ct * 4 + kb) * 32 + lane) << 4));
        v16bf bw1 = *(const v16bf*)(wp1 + (((ct * 4 + kb) * 32 + lane) << 4));
        acc0 = __builtin_amdgcn_wmma_f32_16x16x32_bf16(
                   false, a, false, bw0, (short)0, acc0, false, false);
        acc1 = __builtin_amdgcn_wmma_f32_16x16x32_bf16(
                   false, a, false, bw1, (short)0, acc1, false, false);
    }

    // C/D layout: VGPR j -> lanes 0-15: M=j, lanes 16-31: M=j+8; N = lane&15
    const int   n     = ct * 16 + (lane & 15);
    const int   mrow  = rowbase + khalf * 8;
    const float bias0 = b0[n];
    const float bias1 = b1[n];
#pragma unroll
    for (int j = 0; j < 8; ++j) {
        size_t off = (size_t)(mrow + j) * F + n;
        y[off] = acc0[j] + bias0;
        m[off] = acc1[j] + bias1;
    }
}

// ---------------------------------------------------------------------------
// Edge aggregation: one wave per edge.
//   y[d] += m[s] ; y[s] += m[d]
// Each lane moves one float4 (32 lanes * 4 = 128 features).
// ---------------------------------------------------------------------------
__global__ __launch_bounds__(256) void edge_agg_kernel(
    const long long* __restrict__ edges,
    const float* __restrict__ m,
    float* __restrict__ y, int nedges)
{
    int e = blockIdx.x * 8 + (threadIdx.x >> 5);
    if (e >= nedges) return;
    int lane = threadIdx.x & 31;

    long long s = edges[2 * (size_t)e];
    long long d = edges[2 * (size_t)e + 1];

    const float4 vs = ((const float4*)(m + (size_t)s * F))[lane];
    const float4 vd = ((const float4*)(m + (size_t)d * F))[lane];

    float* yd = y + (size_t)d * F + lane * 4;
    float* ys = y + (size_t)s * F + lane * 4;
    atomAddF(yd + 0, vs.x); atomAddF(yd + 1, vs.y);
    atomAddF(yd + 2, vs.z); atomAddF(yd + 3, vs.w);
    atomAddF(ys + 0, vd.x); atomAddF(ys + 1, vd.y);
    atomAddF(ys + 2, vd.z); atomAddF(ys + 3, vd.w);
}

// ---------------------------------------------------------------------------
// BatchNorm stats: column sums / sums-of-squares (128 threads = 128 columns)
// ---------------------------------------------------------------------------
__global__ __launch_bounds__(128) void zero_stats_kernel(float* stats) {
    // 256 floats (sum + sumsq), 128 threads -> 2 each
    stats[threadIdx.x] = 0.0f;
    stats[threadIdx.x + 128] = 0.0f;
}

__global__ __launch_bounds__(128) void bn_stats_kernel(const float* __restrict__ y,
                                                       float* __restrict__ colsum,
                                                       float* __restrict__ colsumsq,
                                                       int n) {
    int c = threadIdx.x;
    float s = 0.0f, s2 = 0.0f;
    for (int r = blockIdx.x; r < n; r += gridDim.x) {
        float v = y[(size_t)r * F + c];
        s  += v;
        s2 += v * v;
    }
    atomAddF(&colsum[c], s);
    atomAddF(&colsumsq[c], s2);
}

// ---------------------------------------------------------------------------
// BatchNorm apply + ReLU + convert to bf16 for next layer's WMMA input
// ---------------------------------------------------------------------------
__global__ __launch_bounds__(256) void bn_apply_kernel(
    const float* __restrict__ y,
    const float* __restrict__ colsum, const float* __restrict__ colsumsq,
    const float* __restrict__ gamma,  const float* __restrict__ beta,
    bf16_t* __restrict__ xb, int total)
{
    int idx = blockIdx.x * blockDim.x + threadIdx.x;
    if (idx >= total) return;
    int c = idx & (F - 1);
    const float inv_n = 1.0f / (float)NND;
    float mean = colsum[c] * inv_n;
    float var  = colsumsq[c] * inv_n - mean * mean;
    float v = (y[idx] - mean) * rsqrtf(var + EPSV) * gamma[c] + beta[c];
    v = fmaxf(v, 0.0f);
    xb[idx] = (bf16_t)v;
}

// ---------------------------------------------------------------------------
// Host-side launch
// ---------------------------------------------------------------------------
extern "C" void kernel_launch(void* const* d_in, const int* in_sizes, int n_in,
                              void* d_out, int out_size, void* d_ws, size_t ws_size,
                              hipStream_t stream) {
    const float*     features = (const float*)d_in[0];
    const long long* edges    = (const long long*)d_in[1];

    // params flattened per layer (sorted dict keys): b0, b1, beta, gamma, w0, w1
    const float* b0[4]; const float* b1[4];
    const float* betaP[4]; const float* gammaP[4];
    const float* w0[4]; const float* w1[4];
    for (int l = 0; l < 4; ++l) {
        int base = 2 + 6 * l;
        b0[l]    = (const float*)d_in[base + 0];
        b1[l]    = (const float*)d_in[base + 1];
        betaP[l] = (const float*)d_in[base + 2];
        gammaP[l]= (const float*)d_in[base + 3];
        w0[l]    = (const float*)d_in[base + 4];
        w1[l]    = (const float*)d_in[base + 5];
    }

    // workspace carve-up (256B aligned regions)
    char* ws = (char*)d_ws;
    size_t off = 0;
    auto carve = [&](size_t bytes) -> char* {
        char* p = ws + off;
        off = (off + bytes + 255) & ~(size_t)255;
        return p;
    };
    bf16_t* xb   = (bf16_t*)carve((size_t)NND * F * sizeof(bf16_t)); // 25.6 MB
    float*  ybuf = (float*) carve((size_t)NND * F * sizeof(float));  // 51.2 MB
    float*  mbuf = (float*) carve((size_t)NND * F * sizeof(float));  // 51.2 MB
    bf16_t* wp   = (bf16_t*)carve((size_t)8 * F * F * sizeof(bf16_t)); // 256 KB
    float*  stats = (float*)carve(256 * sizeof(float));              // sum | sumsq

    // 1) pack all 8 weight matrices into WMMA B-fragment layout
    for (int l = 0; l < 4; ++l) {
        pack_w_kernel<<<(F * F + 255) / 256, 256, 0, stream>>>(
            w0[l], wp + (size_t)(2 * l) * F * F);
        pack_w_kernel<<<(F * F + 255) / 256, 256, 0, stream>>>(
            w1[l], wp + (size_t)(2 * l + 1) * F * F);
    }

    // 2) features -> bf16
    const int total = NND * F;
    cvt_kernel<<<(total + 255) / 256, 256, 0, stream>>>(features, xb, total);

    // 3) layers
    const int gemm_blocks = NND / 16;          // 6250 (exact)
    const int edge_blocks = NEDG / 8;          // 200000 (exact, 8 edges/block)
    for (int l = 0; l < 4; ++l) {
        float* yout = (l == 3) ? (float*)d_out : ybuf;
        const bf16_t* wpl0 = wp + (size_t)(2 * l) * F * F;
        const bf16_t* wpl1 = wp + (size_t)(2 * l + 1) * F * F;

        gemm_dual_kernel<<<gemm_blocks, 256, 0, stream>>>(
            xb, wpl0, b0[l], wpl1, b1[l], yout, mbuf);

        edge_agg_kernel<<<edge_blocks, 256, 0, stream>>>(edges, mbuf, yout, NEDG);

        if (l < 3) {
            zero_stats_kernel<<<1, 128, 0, stream>>>(stats);
            bn_stats_kernel<<<512, 128, 0, stream>>>(yout, stats, stats + 128, NND);
            bn_apply_kernel<<<(total + 255) / 256, 256, 0, stream>>>(
                yout, stats, stats + 128, gammaP[l], betaP[l], xb, total);
        }
    }
}
